// assert_model_21930103013598
// MI455X (gfx1250) — matrix operand order repaired and checked by reading, not compiled, not run on hardware
//
#include <hip/hip_runtime.h>
#include <hip/hip_bf16.h>

typedef __attribute__((ext_vector_type(16))) __bf16 v16bf;
typedef __attribute__((ext_vector_type(8)))  __bf16 v8bf;
typedef __attribute__((ext_vector_type(8)))  float  v8f;
typedef __attribute__((ext_vector_type(4)))  float  v4f;

// GPT-2 small structural constants
#define LNUM  12
#define HNUM  12
#define DMOD  768
#define DHEAD 64
#define FDIM  3072
#define VOCAB 50257
#define SEQ   128
#define NCIR  29
#define SD    (SEQ*DMOD)    // 98304
#define SDH   (SEQ*DHEAD)   // 8192
#define SSQ   (SEQ*SEQ)     // 16384
#define SF    (SEQ*FDIM)    // 393216

// ------------------------------------------------------------------
// Generic batched GEMM: C[M,N] = A[M,K] * B(+T)[K,N] (+bias)(+gelu)
// bf16 WMMA (v_wmma_f32_16x16x32_bf16), f32 accumulate.
// flags: bit0 = B transposed (B given as [N,K]), bit1 = bias, bit2 = gelu
// Requirements: M,K multiples of 32; for NN, N multiple of 64 (ragged N
// is only used on the NT path: logits @ wte^T).
// ------------------------------------------------------------------
#define BM 128
#define BN 64
#define BK 32
#define KPAD 40   // padded k-stride (elements) for LDS rows; 80B keeps 16B align

__device__ __forceinline__ float gelu_new(float x) {
  float u = 0.7978845608028654f * (x + 0.044715f * x * x * x);
  return 0.5f * x * (1.0f + tanhf(u));
}

__global__ __launch_bounds__(256) void gemm_wmma_kernel(
    int M, int N, int K,
    const float* __restrict__ A, int lda, long long sA,
    const float* __restrict__ B, int ldb, long long sB,
    float* __restrict__ C, int ldc, long long sC,
    const float* __restrict__ bias, long long sBias, int flags)
{
  __shared__ __bf16 As[BM][KPAD];   // [m][k]
  __shared__ __bf16 Bs[BN][KPAD];   // [n][k]
  const int bz = blockIdx.z;
  A += (long long)bz * sA;
  B += (long long)bz * sB;
  C += (long long)bz * sC;
  const float* biasp = bias ? (bias + (long long)bz * sBias) : nullptr;
  const int n0 = blockIdx.x * BN;
  const int m0 = blockIdx.y * BM;
  const int tid  = (int)threadIdx.x;
  const int lane = tid & 31;
  const int w    = tid >> 5;     // 0..7 waves
  const int wm   = w >> 1;       // 0..3 -> 32-row strip
  const int wn   = w & 1;        // 0..1 -> 32-col strip
  const int lm   = lane & 15;
  const int l16  = lane >> 4;
  const bool bt  = (flags & 1) != 0;

  v8f acc[2][2];
  #pragma unroll
  for (int a = 0; a < 2; ++a)
    #pragma unroll
    for (int c = 0; c < 2; ++c)
      #pragma unroll
      for (int e = 0; e < 8; ++e) acc[a][c][e] = 0.0f;

  // staging-thread constants
  const int arow = tid >> 1;            // A: 0..127
  const int akh  = (tid & 1) * 16;      // A: 0 or 16
  const int nn_n  = tid & 63;           // B NN: col in tile
  const int nn_kq = tid >> 6;           // B NN: 0..3 (8 k each)
  const int nt_n  = tid >> 2;           // B NT: 0..63
  const int nt_kh = (tid & 3) * 8;      // B NT: 0,8,16,24
  const int nt_gc = n0 + nt_n;
  const int nt_nc = (nt_gc < N) ? nt_gc : (N - 1);   // clamped (N>=1 always)
  const float nt_mask = (nt_gc < N) ? 1.0f : 0.0f;

  for (int kk = 0; kk < K; kk += BK) {
    { // A tile 128x32: 16 consecutive f32 -> two packed v8bf stores
      const v4f* src = (const v4f*)(A + (long long)(m0 + arow) * lda + kk + akh);
      v4f f0 = src[0], f1 = src[1], f2 = src[2], f3 = src[3];
      v8bf p0, p1;
      #pragma unroll
      for (int e = 0; e < 4; ++e) {
        p0[e]     = (__bf16)f0[e];
        p0[e + 4] = (__bf16)f1[e];
        p1[e]     = (__bf16)f2[e];
        p1[e + 4] = (__bf16)f3[e];
      }
      *(v8bf*)&As[arow][akh]     = p0;
      *(v8bf*)&As[arow][akh + 8] = p1;
    }
    if (!bt) { // B row-major [K,N]; N multiple of 64 -> no bounds check
      const float* bp = B + (long long)(kk + nn_kq * 8) * ldb + (n0 + nn_n);
      v8bf pk;
      #pragma unroll
      for (int r = 0; r < 8; ++r) pk[r] = (__bf16)bp[(long long)r * ldb];
      *(v8bf*)&Bs[nn_n][nn_kq * 8] = pk;
    } else {   // B is [N,K]; clamped address + mask (branch-free edge)
      const v4f* src = (const v4f*)(B + (long long)nt_nc * ldb + kk + nt_kh);
      v4f f0 = src[0], f1 = src[1];
      v8bf pk;
      #pragma unroll
      for (int e = 0; e < 4; ++e) {
        pk[e]     = (__bf16)(f0[e] * nt_mask);
        pk[e + 4] = (__bf16)(f1[e] * nt_mask);
      }
      *(v8bf*)&Bs[nt_n][nt_kh] = pk;
    }
    __syncthreads();

    // Fragments use contiguous k order (k = 16*l16 + e), same in A and B:
    // A 16x32: lane<16 holds K 0..15 of row lm; lane>=16 holds K 16..31
    // B 32x16: lane<16 holds K 0..15 of col lm; lane>=16 holds K 16..31
    v16bf af[2], bfr[2];
    #pragma unroll
    for (int mt = 0; mt < 2; ++mt) {
      int row = wm * 32 + mt * 16 + lm;
      int kb  = l16 * 16;
      #pragma unroll
      for (int e = 0; e < 16; ++e) {
        af[mt][e]     = As[row][kb + e];
      }
    }
    #pragma unroll
    for (int nt = 0; nt < 2; ++nt) {
      int col = wn * 32 + nt * 16 + lm;
      int kb  = l16 * 16;
      #pragma unroll
      for (int e = 0; e < 16; ++e) bfr[nt][e] = Bs[col][kb + e];
    }
    #pragma unroll
    for (int mt = 0; mt < 2; ++mt)
      #pragma unroll
      for (int nt = 0; nt < 2; ++nt)
        acc[mt][nt] = __builtin_amdgcn_wmma_f32_16x16x32_bf16(
            false, af[mt], false, bfr[nt], (short)0, acc[mt][nt], false, false);
    __syncthreads();
  }

  const bool hasBias = (flags & 2) != 0;
  const bool doGelu  = (flags & 4) != 0;
  #pragma unroll
  for (int mt = 0; mt < 2; ++mt)
    #pragma unroll
    for (int nt = 0; nt < 2; ++nt) {
      int col = n0 + wn * 32 + nt * 16 + lm;
      if (col >= N) continue;
      float bv = hasBias ? biasp[col] : 0.0f;
      #pragma unroll
      for (int v = 0; v < 8; ++v) {      // D layout: VGPR v -> row v + 8*l16
        int row = m0 + wm * 32 + mt * 16 + v + 8 * l16;
        if (row >= M) continue;
        float x = acc[mt][nt][v] + bv;
        if (doGelu) x = gelu_new(x);
        C[(long long)row * ldc + col] = x;
      }
    }
}

// ------------------------------------------------------------------
// LayerNorm over D=768, one block (256 thr) per row; in-place safe.
// ------------------------------------------------------------------
__global__ __launch_bounds__(256) void ln_kernel(
    const float* __restrict__ in, float* __restrict__ out,
    const float* __restrict__ g, const float* __restrict__ b)
{
  __shared__ float red[256];
  long long row = blockIdx.x;
  const float* x = in + row * DMOD;
  float* y = out + row * DMOD;
  int t = (int)threadIdx.x;
  float v0 = x[t], v1 = x[t + 256], v2 = x[t + 512];
  red[t] = v0 + v1 + v2; __syncthreads();
  for (int o = 128; o; o >>= 1) { if (t < o) red[t] += red[t + o]; __syncthreads(); }
  float m = red[0] * (1.0f / 768.0f);
  __syncthreads();
  float d0 = v0 - m, d1 = v1 - m, d2 = v2 - m;
  red[t] = d0 * d0 + d1 * d1 + d2 * d2; __syncthreads();
  for (int o = 128; o; o >>= 1) { if (t < o) red[t] += red[t + o]; __syncthreads(); }
  float rs = rsqrtf(red[0] * (1.0f / 768.0f) + 1e-5f);
  y[t]       = d0 * rs * g[t]       + b[t];
  y[t + 256] = d1 * rs * g[t + 256] + b[t + 256];
  y[t + 512] = d2 * rs * g[t + 512] + b[t + 512];
}

// ------------------------------------------------------------------
// Causal softmax over S=128 columns, scale 1/sqrt(64); row = h*S+s.
// ------------------------------------------------------------------
__global__ __launch_bounds__(128) void softmax_kernel(float* __restrict__ sc)
{
  __shared__ float red[128];
  int row = blockIdx.x;            // 0 .. H*S-1
  int s = row & (SEQ - 1);
  float* p = sc + (long long)row * SEQ;
  int t = (int)threadIdx.x;
  float v = (t <= s) ? p[t] * 0.125f : -3.0e38f;
  red[t] = v; __syncthreads();
  for (int o = 64; o; o >>= 1) { if (t < o) red[t] = fmaxf(red[t], red[t + o]); __syncthreads(); }
  float m = red[0]; __syncthreads();
  float e = __expf(v - m);
  red[t] = e; __syncthreads();
  for (int o = 64; o; o >>= 1) { if (t < o) red[t] += red[t + o]; __syncthreads(); }
  p[t] = e / red[0];
}

// ------------------------------------------------------------------
// Embedding: h[s,:] = wte[ids[s],:] + wpe[s,:]
// ------------------------------------------------------------------
__global__ __launch_bounds__(256) void embed_kernel(
    const int* __restrict__ ids, const float* __restrict__ wte,
    const float* __restrict__ wpe, float* __restrict__ h)
{
  int s = blockIdx.x, t = (int)threadIdx.x;
  long long id = ids[s];
  #pragma unroll
  for (int j = 0; j < 3; ++j) {
    int d = t + j * 256;
    h[(long long)s * DMOD + d] = wte[id * DMOD + d] + wpe[(long long)s * DMOD + d];
  }
}

// ------------------------------------------------------------------
// Checks for layer i: 26 outputs = h - RM[base+r,:] . records[0:K]
// r=0 -> records[i,0] (c1); r=1..12 -> x2; r=13 -> x3; r=14..25 -> x4
// K = i*NCIR (K=0 => pure broadcast of h, layer 0 case)
// ------------------------------------------------------------------
__global__ __launch_bounds__(256) void check_kernel(
    const float* h, const float* records, const int* __restrict__ rm,
    int baseRow, int K,
    float* rec_c1, float* x2, float* x3, float* x4)
{
  __shared__ float srm[26 * (LNUM * NCIR)];
  int tid = (int)threadIdx.x;
  for (int i = tid; i < 26 * K; i += 256) {
    int r = i / K, j = i - r * K;
    srm[r * K + j] = (float)rm[(long long)(baseRow + r) * (LNUM * NCIR) + j];
  }
  __syncthreads();
  long long idx = (long long)blockIdx.x * 256 + tid;
  float base = h[idx];
  float acc[26];
  #pragma unroll
  for (int r = 0; r < 26; ++r) acc[r] = base;
  for (int j = 0; j < K; ++j) {
    float v = records[(long long)j * SD + idx];
    #pragma unroll
    for (int r = 0; r < 26; ++r) acc[r] -= srm[r * K + j] * v;
  }
  rec_c1[idx] = acc[0];
  #pragma unroll
  for (int hh = 0; hh < 12; ++hh) x2[(long long)hh * SD + idx] = acc[1 + hh];
  x3[idx] = acc[13];
  #pragma unroll
  for (int hh = 0; hh < 12; ++hh) x4[(long long)hh * SD + idx] = acc[14 + hh];
}

// residual = c1 + sum_h head2[h] + proj_b   (reads records slots 0..12)
__global__ __launch_bounds__(256) void residual_kernel(
    const float* rec, const float* __restrict__ pb, float* __restrict__ res)
{
  long long idx = (long long)blockIdx.x * 256 + threadIdx.x;
  int d = (int)(idx % DMOD);
  float v = rec[idx];
  #pragma unroll
  for (int h = 0; h < 12; ++h) v += rec[(long long)(1 + h) * SD + idx];
  res[idx] = v + pb[d];
}

// c5 = stream - c3 - sum_h c4[h]; write c5/c6/ob records; h = residual+stream+mb
__global__ __launch_bounds__(256) void finalize_kernel(
    float* rec, const float* __restrict__ streamv, const float* __restrict__ residual,
    const float* __restrict__ mb, const float* __restrict__ pb, float* __restrict__ hbuf)
{
  long long idx = (long long)blockIdx.x * 256 + threadIdx.x;
  int d = (int)(idx % DMOD);
  float c5 = streamv[idx] - rec[13LL * SD + idx];
  #pragma unroll
  for (int h = 0; h < 12; ++h) c5 -= rec[(long long)(14 + h) * SD + idx];
  rec[26LL * SD + idx] = c5;
  rec[27LL * SD + idx] = mb[d];
  rec[28LL * SD + idx] = pb[d];
  hbuf[idx] = residual[idx] + streamv[idx] + mb[d];
}

// ------------------------------------------------------------------
// Host orchestration
// ------------------------------------------------------------------
static void launch_gemm(hipStream_t st, int M, int N, int K,
                        const float* A, int lda, long long sA,
                        const float* B, int ldb, long long sB,
                        float* C, int ldc, long long sC,
                        const float* bias, long long sBias, int flags, int batch)
{
  dim3 grid((N + BN - 1) / BN, (M + BM - 1) / BM, batch);
  gemm_wmma_kernel<<<grid, 256, 0, st>>>(M, N, K, A, lda, sA, B, ldb, sB,
                                         C, ldc, sC, bias, sBias, flags);
}

// per-head attention branch: xln[H,S,D] -> outC (H tiles of [S,D], stride outStride)
static void run_attention(hipStream_t st, const float* xln,
                          const float* qw, const float* qb, const float* pw,
                          float* q, float* k, float* v, float* sc, float* ctx,
                          float* outC, long long outStride)
{
  // q/k/v: [S,768]@[768,64] per head; W*h = qkv_w[:, p*768 + h*64 .. +64]
  launch_gemm(st, SEQ, DHEAD, DMOD, xln, DMOD, SD, qw + 0,    3 * DMOD, DHEAD, q, DHEAD, SDH, qb + 0,        DHEAD, 2, HNUM);
  launch_gemm(st, SEQ, DHEAD, DMOD, xln, DMOD, SD, qw + 768,  3 * DMOD, DHEAD, k, DHEAD, SDH, qb + DMOD,     DHEAD, 2, HNUM);
  launch_gemm(st, SEQ, DHEAD, DMOD, xln, DMOD, SD, qw + 1536, 3 * DMOD, DHEAD, v, DHEAD, SDH, qb + 2 * DMOD, DHEAD, 2, HNUM);
  // scores = q @ k^T   (NT)
  launch_gemm(st, SEQ, SEQ, DHEAD, q, DHEAD, SDH, k, DHEAD, SDH, sc, SEQ, SSQ, nullptr, 0, 1, HNUM);
  softmax_kernel<<<HNUM * SEQ, 128, 0, st>>>(sc);
  // ctx = attn @ v
  launch_gemm(st, SEQ, DHEAD, SEQ, sc, SEQ, SSQ, v, DHEAD, SDH, ctx, DHEAD, SDH, nullptr, 0, 0, HNUM);
  // out = ctx @ Woh (proj_w rows h*64..h*64+64)
  launch_gemm(st, SEQ, DMOD, DHEAD, ctx, DHEAD, SDH, pw, DMOD, (long long)DHEAD * DMOD,
              outC, DMOD, outStride, nullptr, 0, 0, HNUM);
}

extern "C" void kernel_launch(void* const* d_in, const int* in_sizes, int n_in,
                              void* d_out, int out_size, void* d_ws, size_t ws_size,
                              hipStream_t stream)
{
  (void)in_sizes; (void)n_in; (void)out_size; (void)ws_size;
  const int*   input_ids = (const int*)d_in[0];
  const int*   rm        = (const int*)d_in[2];
  const float* wte       = (const float*)d_in[3];
  const float* wpe       = (const float*)d_in[4];
  const float* qkv_w     = (const float*)d_in[5];
  const float* qkv_b     = (const float*)d_in[6];
  const float* proj_w    = (const float*)d_in[7];
  const float* proj_b    = (const float*)d_in[8];
  const float* ln1_g     = (const float*)d_in[9];
  const float* ln1_b     = (const float*)d_in[10];
  const float* ln2_g     = (const float*)d_in[11];
  const float* ln2_b     = (const float*)d_in[12];
  const float* fc_w      = (const float*)d_in[13];
  const float* fc_b      = (const float*)d_in[14];
  const float* mproj_w   = (const float*)d_in[15];
  const float* mproj_b   = (const float*)d_in[16];
  const float* lnf_g     = (const float*)d_in[17];
  const float* lnf_b     = (const float*)d_in[18];
  float* out = (float*)d_out;

  // workspace layout (floats)
  float* p = (float*)d_ws;
  float* records = p; p += (size_t)LNUM * NCIR * SD;   // 34,209,792
  float* x2      = p; p += (size_t)HNUM * SD;
  float* x4      = p; p += (size_t)HNUM * SD;
  float* x2ln    = p; p += (size_t)HNUM * SD;
  float* x4ln    = p; p += (size_t)HNUM * SD;
  float* head4   = p; p += (size_t)HNUM * SD;
  float* x3t     = p; p += SD;
  float* q       = p; p += (size_t)HNUM * SDH;
  float* k       = p; p += (size_t)HNUM * SDH;
  float* v       = p; p += (size_t)HNUM * SDH;
  float* ctx     = p; p += (size_t)HNUM * SDH;
  float* sc      = p; p += (size_t)HNUM * SSQ;
  float* residual= p; p += SD;
  float* rln     = p; p += SD;
  float* streamb = p; p += SD;
  float* hbuf    = p; p += SD;
  float* hln     = p; p += SD;
  float* act1    = p; p += SF;
  float* act4    = p; p += (size_t)HNUM * SF;

  embed_kernel<<<SEQ, 256, 0, stream>>>(input_ids, wte, wpe, hbuf);

  for (int i = 0; i < LNUM; ++i) {
    const float* qw  = qkv_w   + (size_t)i * DMOD * 3 * DMOD;
    const float* qb  = qkv_b   + (size_t)i * 3 * DMOD;
    const float* pw  = proj_w  + (size_t)i * DMOD * DMOD;
    const float* pb  = proj_b  + (size_t)i * DMOD;
    const float* l1g = ln1_g   + (size_t)i * DMOD;
    const float* l1b = ln1_b   + (size_t)i * DMOD;
    const float* l2g = ln2_g   + (size_t)i * DMOD;
    const float* l2b = ln2_b   + (size_t)i * DMOD;
    const float* fw  = fc_w    + (size_t)i * DMOD * FDIM;
    const float* fb  = fc_b    + (size_t)i * FDIM;
    const float* mw  = mproj_w + (size_t)i * FDIM * DMOD;
    const float* mb  = mproj_b + (size_t)i * DMOD;
    float* recL = records + (size_t)i * NCIR * SD;

    // checks (layer 0: K=0 -> broadcast of h)
    check_kernel<<<SD / 256, 256, 0, stream>>>(hbuf, records, rm, i * NCIR, i * NCIR,
                                               recL, x2, x3t, x4);
    // layernorms of checked inputs
    ln_kernel<<<HNUM * SEQ, 256, 0, stream>>>(x2, x2ln, l1g, l1b);
    ln_kernel<<<HNUM * SEQ, 256, 0, stream>>>(x4, x4ln, l1g, l1b);
    ln_kernel<<<SEQ, 256, 0, stream>>>(x3t, x3t, l1g, l1b);   // in-place double LN
    ln_kernel<<<SEQ, 256, 0, stream>>>(x3t, x3t, l2g, l2b);

    // circuit 2: per-head attention -> records slots 1..12
    run_attention(stream, x2ln, qw, qb, pw, q, k, v, sc, ctx, recL + SD, SD);

    // residual = c1 + sum(head2) + proj_b
    residual_kernel<<<SD / 256, 256, 0, stream>>>(recL, pb, residual);

    // stream = gelu(ln2(residual)@fc + fb) @ mproj
    ln_kernel<<<SEQ, 256, 0, stream>>>(residual, rln, l2g, l2b);
    launch_gemm(stream, SEQ, FDIM, DMOD, rln,  DMOD, 0, fw, FDIM, 0, act1, FDIM, 0, fb, 0, 6, 1);
    launch_gemm(stream, SEQ, DMOD, FDIM, act1, FDIM, 0, mw, DMOD, 0, streamb, DMOD, 0, nullptr, 0, 0, 1);

    // circuit 3: gelu(ln2(ln1(x3))@fc) @ mproj -> records slot 13
    launch_gemm(stream, SEQ, FDIM, DMOD, x3t,  DMOD, 0, fw, FDIM, 0, act1, FDIM, 0, nullptr, 0, 4, 1);
    launch_gemm(stream, SEQ, DMOD, FDIM, act1, FDIM, 0, mw, DMOD, 0, recL + 13LL * SD, DMOD, 0, nullptr, 0, 0, 1);

    // circuit 4: per-head attention -> per-head MLP -> records slots 14..25
    run_attention(stream, x4ln, qw, qb, pw, q, k, v, sc, ctx, head4, SD);
    ln_kernel<<<HNUM * SEQ, 256, 0, stream>>>(head4, head4, l2g, l2b);  // in-place ln2
    launch_gemm(stream, SEQ, FDIM, DMOD, head4, DMOD, SD, fw, FDIM, 0, act4, FDIM, SF, nullptr, 0, 4, HNUM);
    launch_gemm(stream, SEQ, DMOD, FDIM, act4,  FDIM, SF, mw, DMOD, 0, recL + 14LL * SD, DMOD, SD, nullptr, 0, 0, HNUM);

    // c5, c6, ob records + h update
    finalize_kernel<<<SD / 256, 256, 0, stream>>>(recL, streamb, residual, mb, pb, hbuf);
  }

  // logits = ln_f(h) @ wte^T
  ln_kernel<<<SEQ, 256, 0, stream>>>(hbuf, hln, lnf_g, lnf_b);
  launch_gemm(stream, SEQ, VOCAB, DMOD, hln, DMOD, 0, wte, DMOD, 0, out, VOCAB, 0,
              nullptr, 0, 1, 1);
}
